// CRF_91285234909741
// MI455X (gfx1250) — matrix operand order repaired and checked by reading, not compiled, not run on hardware
//
#include <hip/hip_runtime.h>
#include <hip/hip_bf16.h>
#include <math.h>

// CRF forward on MI455X (gfx1250, wave32).
// Log-semiring matmul converted to real f16 GEMM via exp-trick; B-operand
// (exp(transitions)) resident in VGPRs for the whole 1024-step scan.
// Emissions streamed via async global->LDS DMA (ASYNCcnt), issued as inline
// asm (the clang builtin wants un-spellable CUDA-AS typed pointers).

typedef __attribute__((ext_vector_type(16))) _Float16 v16h;
typedef __attribute__((ext_vector_type(8)))  _Float16 v8h;
typedef __attribute__((ext_vector_type(8)))  float    v8f;

#define CRF_B 128
#define CRF_S 1024
#define CRF_K 256
#define BPG   16      // batch rows per workgroup (== WMMA M)
#define SCP   264     // padded score row stride (floats)  -> conflict-free
#define PPD   264     // padded P row stride (halves)      -> conflict-free A loads

// Raw hardware transcendentals: inputs here can never be denormal, so skip
// the libm denorm guard code (v_cmp/v_cndmask/v_ldexp clusters).
#if __has_builtin(__builtin_amdgcn_exp2f) && __has_builtin(__builtin_amdgcn_logf)
__device__ __forceinline__ float fexp(float x) { return __builtin_amdgcn_exp2f(x * 1.4426950408889634f); }
__device__ __forceinline__ float flog(float x) { return __builtin_amdgcn_logf(x) * 0.6931471805599453f; }
#else
__device__ __forceinline__ float fexp(float x) { return __expf(x); }
__device__ __forceinline__ float flog(float x) { return __logf(x); }
#endif

// Async global->LDS staging (gfx1250 ASYNCcnt path).
#if defined(__gfx1250__) && __has_builtin(__builtin_amdgcn_s_wait_asynccnt)
#define CRF_ASYNC_STAGE 1
// VFLAT async DMA: vdst = per-lane LDS byte offset, vaddr = 64-bit global
// address (GV mode). Generic pointers to __shared__ are {aperture_hi,
// lds_offset}, so truncating to 32 bits yields the LDS address the HW wants.
__device__ __forceinline__ void crf_async_ld16(const float* gp, float* lp)
{
    const uint32_t lds = (uint32_t)(uintptr_t)lp;
    asm volatile("global_load_async_to_lds_b128 %0, %1, off"
                 :: "v"(lds), "v"(gp)
                 : "memory");
}
#else
#define CRF_ASYNC_STAGE 0
#endif

__global__ __launch_bounds__(256, 1)
void crf_forward_scan(const float* __restrict__ emis,
                      const unsigned char* __restrict__ masks,
                      const float* __restrict__ startT,
                      const float* __restrict__ endT,
                      const float* __restrict__ trans,
                      const float* __restrict__ num,
                      float* __restrict__ out)
{
    __shared__ __attribute__((aligned(16))) float    sc[BPG * SCP];      // f32 scores
    __shared__ __attribute__((aligned(16))) _Float16 pm[BPG * PPD];      // exp(sc - m)
    __shared__ __attribute__((aligned(16))) float    ebuf[2][BPG][CRF_K];// staged emissions
    __shared__ float         mrow[BPG];
    __shared__ unsigned char mbuf[BPG];

    const int tid  = threadIdx.x;
    const int wave = tid >> 5;
    const int lane = tid & 31;
    const int b0   = blockIdx.x * BPG;
    const int lo16 = lane & 15;
    const int hi   = (lane >> 4) & 1;

    // ---- B operand: expT tiles, resident in VGPRs across the whole scan ----
    // B 32x16 in this code's contiguous K order: lane<16 col=lo16 holds K rows kb..kb+15,
    // lanes>=16 hold K rows kb+16..kb+31 (2 halves per VGPR, in order).
    v16h Breg[16];
#pragma unroll
    for (int tile = 0; tile < 2; ++tile) {
        const int bcol = 32 * wave + 16 * tile + lo16;
#pragma unroll
        for (int kk = 0; kk < 8; ++kk) {
            const int kb = 32 * kk + 16 * hi;
            v16h bv;
#pragma unroll
            for (int i = 0; i < 16; ++i)
                bv[i] = (_Float16)fexp(trans[(size_t)(kb + i) * CRF_K + bcol]);
            Breg[tile * 8 + kk] = bv;
        }
    }

    // A-tile fetch matching B's K order: lane<16 row=lo16 elems
    // 0..15 -> K 32kk..+15; lanes>=16 offset +16.
    auto load_a = [&](int kk) -> v16h {
        const int kb0 = 32 * kk + 16 * hi;
        const v8h alo = *reinterpret_cast<const v8h*>(&pm[lo16 * PPD + kb0]);
        const v8h ahi = *reinterpret_cast<const v8h*>(&pm[lo16 * PPD + kb0 + 8]);
        v16h a;
#pragma unroll
        for (int i = 0; i < 8; ++i) { a[i] = alo[i]; a[8 + i] = ahi[i]; }
        return a;
    };

    // ---- score(t=0) = start + emissions[:,0,:]; stage emissions for t=1 ----
#pragma unroll
    for (int i = 0; i < BPG; ++i) {
        const size_t base = (size_t)(b0 + i) * CRF_S * CRF_K;
        sc[i * SCP + tid] = startT[tid] + __builtin_nontemporal_load(&emis[base + tid]);
    }
#if CRF_ASYNC_STAGE
    // async-DMA the t=1 tile: 1024 chunks of 16B, 4 per thread
#pragma unroll
    for (int q = 0; q < 4; ++q) {
        const int c    = tid + 256 * q;
        const int row  = c >> 6;
        const int col4 = (c & 63) << 2;
        crf_async_ld16(&emis[((size_t)(b0 + row) * CRF_S + 1) * CRF_K + col4],
                       &ebuf[1][row][col4]);
    }
#else
#pragma unroll
    for (int i = 0; i < BPG; ++i) {
        const size_t base = (size_t)(b0 + i) * CRF_S * CRF_K;
        ebuf[1][i][tid] = __builtin_nontemporal_load(&emis[base + CRF_K + tid]);
    }
#endif
    __syncthreads();

    for (int t = 1; t < CRF_S; ++t) {
        const int cur = t & 1;
        const int nxt = cur ^ 1;

        // stage emissions for step t+1 (double buffered; overlaps phase A)
        if (t + 1 < CRF_S) {
#if CRF_ASYNC_STAGE
#pragma unroll
            for (int q = 0; q < 4; ++q) {
                const int c    = tid + 256 * q;
                const int row  = c >> 6;
                const int col4 = (c & 63) << 2;
                crf_async_ld16(&emis[((size_t)(b0 + row) * CRF_S + (t + 1)) * CRF_K + col4],
                               &ebuf[nxt][row][col4]);
            }
#else
#pragma unroll
            for (int i = 0; i < BPG; ++i)
                ebuf[nxt][i][tid] = __builtin_nontemporal_load(
                    &emis[((size_t)(b0 + i) * CRF_S + (t + 1)) * CRF_K + tid]);
#endif
        }
        if (tid < BPG) mbuf[tid] = masks[(size_t)(b0 + tid) * CRF_S + t];

        // ---- phase A: rowmax + P = exp(sc - m), two rows per wave ----
#pragma unroll
        for (int r2 = 0; r2 < 2; ++r2) {
            const int row = 2 * wave + r2;
            float v[8];
            float mx = -3.0e38f;
#pragma unroll
            for (int q = 0; q < 8; ++q) {
                v[q] = sc[row * SCP + lane + 32 * q];
                mx = fmaxf(mx, v[q]);
            }
#pragma unroll
            for (int off = 16; off > 0; off >>= 1)
                mx = fmaxf(mx, __shfl_xor(mx, off, 32));
#pragma unroll
            for (int q = 0; q < 8; ++q)
                pm[row * PPD + lane + 32 * q] = (_Float16)fexp(v[q] - mx);
            if (lane == 0) mrow[row] = mx;
        }
#if CRF_ASYNC_STAGE
        // async loads complete in order: waiting <=4 (the 4 just issued for
        // t+1) guarantees the 4 for step t have landed in ebuf[cur].
        if (t + 1 < CRF_S) __builtin_amdgcn_s_wait_asynccnt(4);
        else               __builtin_amdgcn_s_wait_asynccnt(0);
#endif
        __syncthreads();

        // ---- phase B: (16x256) @ (256x256) f16 GEMM, 32 output cols/wave ----
        // 1-deep software pipeline: tile kk+1 loads into fresh registers
        // while the WMMA pair consumes tile kk (keeps DS latency covered).
        v8f acc0 = {};
        v8f acc1 = {};
        v16h a = load_a(0);
#pragma unroll
        for (int kk = 0; kk < 8; ++kk) {
            v16h an = a;
            if (kk < 7) an = load_a(kk + 1);
            acc0 = __builtin_amdgcn_wmma_f32_16x16x32_f16(false, a, false, Breg[kk],
                                                          (short)0, acc0, false, false);
            acc1 = __builtin_amdgcn_wmma_f32_16x16x32_f16(false, a, false, Breg[8 + kk],
                                                          (short)0, acc1, false, false);
            a = an;
        }

        // ---- finish: sc' = m + log(acc) + e[t]; masked update ----
#pragma unroll
        for (int tile = 0; tile < 2; ++tile) {
            const int col = 32 * wave + 16 * tile + lo16;
#pragma unroll
            for (int r = 0; r < 8; ++r) {
                const int row = r + 8 * hi;               // ISA f32 D layout
                const float s = (tile == 0) ? acc0[r] : acc1[r];
                const float v = mrow[row] + flog(s) + ebuf[cur][row][col];
                const float old = sc[row * SCP + col];
                sc[row * SCP + col] = mbuf[row] ? v : old;
            }
        }
        __syncthreads();
    }

    // ---- denominator = lse_j(sc + end); out = num - den ----
#pragma unroll
    for (int r2 = 0; r2 < 2; ++r2) {
        const int row = 2 * wave + r2;
        float v[8];
        float mx = -3.0e38f;
#pragma unroll
        for (int q = 0; q < 8; ++q) {
            const int c = lane + 32 * q;
            v[q] = sc[row * SCP + c] + endT[c];
            mx = fmaxf(mx, v[q]);
        }
#pragma unroll
        for (int off = 16; off > 0; off >>= 1)
            mx = fmaxf(mx, __shfl_xor(mx, off, 32));
        float sum = 0.f;
#pragma unroll
        for (int q = 0; q < 8; ++q) sum += fexp(v[q] - mx);
#pragma unroll
        for (int off = 16; off > 0; off >>= 1)
            sum += __shfl_xor(sum, off, 32);
        if (lane == 0)
            out[b0 + row] = num[b0 + row] - (mx + flog(sum));
    }
}

// Numerator: one wave per batch; gathers + shfl reductions. ~1 MB of traffic.
__global__ __launch_bounds__(256)
void crf_numerator(const float* __restrict__ emis,
                   const int* __restrict__ tgt,
                   const unsigned char* __restrict__ masks,
                   const float* __restrict__ startT,
                   const float* __restrict__ endT,
                   const float* __restrict__ trans,
                   float* __restrict__ num)
{
    const int wave = threadIdx.x >> 5;
    const int lane = threadIdx.x & 31;
    const int b = blockIdx.x * 8 + wave;
    if (b >= CRF_B) return;

    float acc = 0.f;
    int cnt = 0;
    for (int t = lane; t < CRF_S; t += 32) {
        const int tg = tgt[(size_t)b * CRF_S + t];
        const float e = emis[((size_t)b * CRF_S + t) * CRF_K + tg];
        const unsigned char mk = masks[(size_t)b * CRF_S + t];
        cnt += mk ? 1 : 0;
        if (t == 0) {
            acc += startT[tg] + e;                        // t=0 term is unmasked
        } else {
            const int tp = tgt[(size_t)b * CRF_S + t - 1];
            acc += mk ? (trans[(size_t)tp * CRF_K + tg] + e) : 0.f;
        }
    }
#pragma unroll
    for (int off = 16; off > 0; off >>= 1) {
        acc += __shfl_xor(acc, off, 32);
        cnt += __shfl_xor(cnt, off, 32);
    }
    if (lane == 0) {
        const int last = tgt[(size_t)b * CRF_S + (cnt - 1)];
        num[b] = acc + endT[last];
    }
}

extern "C" void kernel_launch(void* const* d_in, const int* in_sizes, int n_in,
                              void* d_out, int out_size, void* d_ws, size_t ws_size,
                              hipStream_t stream)
{
    const float*         emis   = (const float*)d_in[0];
    const int*           tgt    = (const int*)d_in[1];
    const unsigned char* masks  = (const unsigned char*)d_in[2];
    const float*         startT = (const float*)d_in[3];
    const float*         endT   = (const float*)d_in[4];
    const float*         trans  = (const float*)d_in[5];
    float*               out    = (float*)d_out;
    float*               num    = (float*)d_ws;          // 128 floats of scratch

    crf_numerator<<<CRF_B / 8, 256, 0, stream>>>(emis, tgt, masks, startT, endT, trans, num);
    crf_forward_scan<<<CRF_B / BPG, 256, 0, stream>>>(emis, masks, startT, endT, trans, num, out);
}
